// _STFModelBase_47811575939191
// MI455X (gfx1250) — matrix operand order repaired and checked by reading, not compiled, not run on hardware
//
#include <hip/hip_runtime.h>
#include <math.h>

// Problem constants (from reference)
#define D_SETS  256
#define SEG_LEN 65536
#define NCHUNK  16            // chunks per row
#define CHUNK   4096          // elements per chunk (one block)
#define TPB     256           // 8 wave32s

typedef __attribute__((ext_vector_type(16))) _Float16 v16h;
typedef __attribute__((ext_vector_type(8)))  float    v8f;

// g[m] = exclusive-scan feed: g[0]=0, g[m]=stress(t[m-1])*(t[m]-t[m-1])
// so that fit[m] = K0 + inclusive_scan(g)[m].
__device__ __forceinline__ float g_value(const float* __restrict__ xrow, int m,
                                         float a, float b, float cinv, float drpt) {
    if (m == 0) return 0.0f;
    float tp = xrow[m - 1] * 500.0f;   // F in [0,1] -> t in [0,500]
    float tc = xrow[m]     * 500.0f;
    float stress = a * __expf(-tp * cinv) + b * log1pf(fabsf(tp)) + drpt;
    return stress * (tc - tp);
}

__device__ __forceinline__ void row_consts(const float* __restrict__ params,
                                           const float* __restrict__ proc, int r,
                                           float& a, float& b, float& cinv, float& drpt) {
    a = params[4 * r + 0];
    b = params[4 * r + 1];
    float cc = params[4 * r + 2];
    float d  = params[4 * r + 3];
    cinv = 1.0f / (fabsf(cc) + 0.001f);
    float R = proc[4 * r + 0];
    float T = proc[4 * r + 1] + 0.1f;
    float P = proc[4 * r + 2];
    drpt = d * R * P / T;
}

// ---------- Phase 1: per-chunk partial sums of g ----------
__global__ void stf_partials(const float* __restrict__ x,
                             const float* __restrict__ params,
                             const float* __restrict__ proc,
                             float* __restrict__ partials) {
    const int c = blockIdx.x, r = blockIdx.y, tid = threadIdx.x;
    const float* xrow = x + (size_t)r * SEG_LEN;
    float a, b, cinv, drpt;
    row_consts(params, proc, r, a, b, cinv, drpt);

    const int base = c * CHUNK;
    float s = 0.0f;
    for (int k = 0; k < CHUNK / TPB; ++k)            // coalesced sweep
        s += g_value(xrow, base + tid + k * TPB, a, b, cinv, drpt);

    for (int off = 16; off > 0; off >>= 1)           // wave32 reduce
        s += __shfl_down(s, off, 32);

    __shared__ float wsum[TPB / 32];
    const int w = tid >> 5, lane = tid & 31;
    if (lane == 0) wsum[w] = s;
    __syncthreads();
    if (tid == 0) {
        float t = 0.0f;
        for (int i = 0; i < TPB / 32; ++i) t += wsum[i];
        partials[r * NCHUNK + c] = t;
    }
}

// ---------- Phase 2: scan the D x NCHUNK partials, fold in K0 ----------
__global__ void stf_scan_partials(const float* __restrict__ partials,
                                  const float* __restrict__ K0,
                                  float* __restrict__ chunkoff) {
    const int d = threadIdx.x;                       // one thread per row, 256 threads
    float run = K0[d];
    for (int c = 0; c < NCHUNK; ++c) {
        chunkoff[d * NCHUNK + c] = run;
        run += partials[d * NCHUNK + c];
    }
}

// ---------- Phase 3: WMMA tile scans + offsets + normalized store ----------
__global__ void stf_scan_wmma(const float* __restrict__ x,
                              const float* __restrict__ params,
                              const float* __restrict__ proc,
                              const float* __restrict__ chunkoff,
                              float* __restrict__ out) {
    __shared__ float lds[CHUNK];                     // 16 KB staging (of 320 KB LDS)
    __shared__ float wavetot[TPB / 32];

    const int c = blockIdx.x, r = blockIdx.y, tid = threadIdx.x;
    const float* xrow = x + (size_t)r * SEG_LEN;
    float a, b, cinv, drpt;
    row_consts(params, proc, r, a, b, cinv, drpt);
    const int base = c * CHUNK;

    // (a) compute g into LDS, coalesced x reads (phase-3 x re-read hits 192MB L2)
    for (int k = 0; k < CHUNK / TPB; ++k) {
        const int p = tid + k * TPB;
        lds[p] = g_value(xrow, base + p, a, b, cinv, drpt);
    }
    __syncthreads();

    const int w = tid >> 5, lane = tid & 31;
    const int col = lane & 15, h = lane >> 4;        // column index / lane half

    // A = [L | L] (16x32 f16): L lower-triangular ones => Y = L*X_hi + L*X_lo in one WMMA.
    // 16-bit A layout (contiguous, matching B): lane half h, VGPR v, sub j -> K = 16h+2v+j.
    v16h amat;
    for (int v = 0; v < 8; ++v)
        for (int j = 0; j < 2; ++j) {
            const int K = 16 * h + 2 * v + j;
            amat[2 * v + j] = ((K & 15) <= col) ? (_Float16)1.0f : (_Float16)0.0f;
        }

    v8f   y[2];
    float carry = 0.0f;                              // within-wave running sum
    for (int u = 0; u < 2; ++u) {                    // two 256-element units per wave
        // B = [X_hi ; X_lo] (32x16 f16): lanes 0-15 carry hi rows (K=0-15),
        // lanes 16-31 carry lo rows (K=16-31) of the SAME column -> contiguous LDS reads.
        const float* gp = &lds[w * 512 + u * 256 + col * 16];
        v16h bmat;
        for (int s = 0; s < 16; ++s) {
            const float gv = gp[s];
            const _Float16 hi = (_Float16)gv;
            bmat[s] = h ? (_Float16)(gv - (float)hi) : hi;   // hi/lo f16 split
        }

        v8f acc = {};
        acc = __builtin_amdgcn_wmma_f32_16x16x32_f16(
            /*neg_a=*/false, amat, /*neg_b=*/false, bmat,
            /*c_mod=*/(short)0, acc, /*reuse_a=*/false, /*reuse_b=*/false);
        // acc = column-wise inclusive scans (C/D layout: lane l owns column l&15,
        // VGPR v holds rows v (lanes 0-15) / v+8 (lanes 16-31)).

        // Column totals T[j] = Y[15][j] live in VGPR7 of lanes 16-31.
        const float tj = __shfl(acc[7], 16 + col, 32);       // every lane: T of its column
        float incl = tj;                                      // 16-wide Hillis-Steele scan
        for (int dlt = 1; dlt < 16; dlt <<= 1) {
            const float up = __shfl_up(incl, dlt, 16);
            if (col >= dlt) incl += up;
        }
        const float excl  = incl - tj;
        const float utot  = __shfl(incl, 15, 16);            // unit total (broadcast)
        const float add   = excl + carry;
        for (int v = 0; v < 8; ++v) acc[v] += add;
        y[u] = acc;
        carry += utot;
    }

    if (lane == 0) wavetot[w] = carry;                        // wave total over its 512
    __syncthreads();
    float wcarry = 0.0f;
    for (int i = 0; i < w; ++i) wcarry += wavetot[i];

    const float baseoff = chunkoff[r * NCHUNK + c] + wcarry;  // K0 + preceding chunks
    const float inv_rng = 1.0f / 4000.0f;                     // (F_MAX-F_MIN)/(Y_MAX-Y_MIN)

    // Write normalized results back to LDS in scan order (safe: all g reads are
    // before the barrier above), then do a fully coalesced global store.
    for (int u = 0; u < 2; ++u)
        for (int v = 0; v < 8; ++v) {
            const float fit = y[u][v] + baseoff;
            const int   i   = v + 8 * h;                      // matrix row
            lds[w * 512 + u * 256 + col * 16 + i] = (fit + 2000.0f) * inv_rng;
        }
    __syncthreads();

    const size_t gbase = (size_t)r * SEG_LEN + base;
    for (int k = 0; k < CHUNK / TPB; ++k) {
        const int p = tid + k * TPB;
        out[gbase + p] = lds[p];
    }
}

extern "C" void kernel_launch(void* const* d_in, const int* in_sizes, int n_in,
                              void* d_out, int out_size, void* d_ws, size_t ws_size,
                              hipStream_t stream) {
    const float* x      = (const float*)d_in[0];   // [D*L]
    const float* params = (const float*)d_in[1];   // [D,4]
    const float* K0     = (const float*)d_in[2];   // [D]
    const float* proc   = (const float*)d_in[3];   // [D,4]
    float* out = (float*)d_out;

    float* partials = (float*)d_ws;                         // D*NCHUNK floats (16 KB)
    float* chunkoff = partials + D_SETS * NCHUNK;           // D*NCHUNK floats (16 KB)

    dim3 grid(NCHUNK, D_SETS);                              // 4096 blocks x 256 threads
    stf_partials<<<grid, TPB, 0, stream>>>(x, params, proc, partials);
    stf_scan_partials<<<1, D_SETS, 0, stream>>>(partials, K0, chunkoff);
    stf_scan_wmma<<<grid, TPB, 0, stream>>>(x, params, proc, chunkoff, out);
}
